// NENNClassifier_33380485824564
// MI455X (gfx1250) — compile-verified, matrix operand order checked by reading, not run on hardware
//
#include <hip/hip_runtime.h>
#include <hip/hip_bf16.h>

#define HID   64
#define BATCH 64
#define N0    1024
#define E0    8192
#define NN    (BATCH * N0)      // 65536 nodes
#define NE    (BATCH * E0)      // 524288 edges
#define NE3   (3 * NE)          // 1572864 e2e entries
#define NCLS  10
#define LEAK  0.2f

typedef __bf16 bf16;
typedef bf16  bf16x16 __attribute__((ext_vector_type(16)));
typedef float f32x8   __attribute__((ext_vector_type(8)));

// ---------------------------------------------------------------------------
// WMMA GEMM: C[M x 64] = A[M x K] @ W[K x 64], all f32 row-major in memory,
// converted to bf16 in-register. One wave per 16-row strip; A fragment loaded
// with float4 (global_load_b128) and reused across the 4 column tiles.
// ---------------------------------------------------------------------------
__global__ __launch_bounds__(32)
void gemm_mx64_bf16(const float* __restrict__ A, const float* __restrict__ W,
                    float* __restrict__ C, int K, int ldc, int coff) {
    const int lane = threadIdx.x & 31;
    const int half = lane >> 4;
    const int l16  = lane & 15;
    const int m0   = blockIdx.x * 16;

    f32x8 acc[4] = {};
    for (int kc = 0; kc < K; kc += 32) {
        // A fragment: row m0+l16; lanes<16 hold K {0..7,16..23}, lanes>=16 {8..15,24..31}
        const float* arow = A + (size_t)(m0 + l16) * K + kc + half * 8;
        float4 a0 = *(const float4*)(arow);
        float4 a1 = *(const float4*)(arow + 4);
        float4 a2 = *(const float4*)(arow + 16);
        float4 a3 = *(const float4*)(arow + 20);
        bf16x16 a;
        a[0]  = (bf16)a0.x; a[1]  = (bf16)a0.y; a[2]  = (bf16)a0.z; a[3]  = (bf16)a0.w;
        a[4]  = (bf16)a1.x; a[5]  = (bf16)a1.y; a[6]  = (bf16)a1.z; a[7]  = (bf16)a1.w;
        a[8]  = (bf16)a2.x; a[9]  = (bf16)a2.y; a[10] = (bf16)a2.z; a[11] = (bf16)a2.w;
        a[12] = (bf16)a3.x; a[13] = (bf16)a3.y; a[14] = (bf16)a3.z; a[15] = (bf16)a3.w;
#pragma unroll
        for (int nt = 0; nt < 4; ++nt) {
            // B fragment: column nt*16+l16 of W, same K striping as A (W is tiny, L2-hot)
            bf16x16 b;
            const float* wcol = W + (size_t)kc * 64 + nt * 16 + l16;
#pragma unroll
            for (int e = 0; e < 8; ++e) {
                b[e]     = (bf16)wcol[(half * 8 + e) * 64];
                b[e + 8] = (bf16)wcol[(16 + half * 8 + e) * 64];
            }
            acc[nt] = __builtin_amdgcn_wmma_f32_16x16x32_bf16(
                false, a, false, b, (short)0, acc[nt], false, false);
        }
    }
    // C/D layout: VGPR r, lanes 0-15 -> M = r, lanes 16-31 -> M = 8+r, N = l16
#pragma unroll
    for (int nt = 0; nt < 4; ++nt)
#pragma unroll
        for (int r = 0; r < 8; ++r)
            C[(size_t)(m0 + half * 8 + r) * ldc + coff + nt * 16 + l16] = acc[nt][r];
}

// ---------------------------------------------------------------------------
// Helpers / elementwise kernels
// ---------------------------------------------------------------------------
__device__ __forceinline__ float lrelu(float v) { return v > 0.f ? v : LEAK * v; }

__device__ __forceinline__ unsigned enc_f32(float f) {
    unsigned u = __float_as_uint(f);
    return (u & 0x80000000u) ? ~u : (u | 0x80000000u);
}
__device__ __forceinline__ float dec_f32(unsigned u) {
    return (u & 0x80000000u) ? __uint_as_float(u & 0x7FFFFFFFu) : __uint_as_float(~u);
}
#define ENC_NEGINF 0x007FFFFFu   // enc_f32(-inf)

__global__ void fill_u32(unsigned* __restrict__ p, unsigned v, int n) {
    int i = blockIdx.x * blockDim.x + threadIdx.x;
    if (i < n) p[i] = v;
}

// o[i] = dot(h[i, 0:64], v[0:64])
__global__ void dot64(const float* __restrict__ h, const float* __restrict__ v,
                      float* __restrict__ o, int M) {
    int i = blockIdx.x * blockDim.x + threadIdx.x;
    if (i >= M) return;
    const float* row = h + (size_t)i * 64;
    float acc = 0.f;
#pragma unroll
    for (int c = 0; c < 64; ++c) acc += row[c] * v[c];
    o[i] = acc;
}

// s[j] = leaky_relu(q[qidx[j]] + (kidx ? k[kidx[j]] : k[j])); seg-max into m_enc[qidx[j]]
__global__ void score_max(const float* __restrict__ q, const float* __restrict__ k,
                          const int* __restrict__ qidx, const int* __restrict__ kidx,
                          float* __restrict__ s, unsigned* __restrict__ m_enc, int n) {
    int j = blockIdx.x * blockDim.x + threadIdx.x;
    if (j >= n) return;
    int d = qidx[j];
    float kv = kidx ? k[kidx[j]] : k[j];
    float v = lrelu(q[d] + kv);
    s[j] = v;
    atomicMax(&m_enc[d], enc_f32(v));
}

// s[j] = exp(s[j] - m[seg[j]]); z[seg[j]] += s[j]   (m decoded; -inf -> 0 like reference)
__global__ void exp_sum(float* __restrict__ s, const unsigned* __restrict__ m_enc,
                        const int* __restrict__ seg, float* __restrict__ z, int n) {
    int j = blockIdx.x * blockDim.x + threadIdx.x;
    if (j >= n) return;
    int d = seg[j];
    float m = dec_f32(m_enc[d]);
    if (!isfinite(m)) m = 0.f;
    float e = __expf(s[j] - m);
    s[j] = e;
    atomicAdd(&z[d], e);
}

// out[seg[j]*ldo + ooff + c] += (s[j]/(z[seg[j]]+1e-16)) * h[(gat?gat[j]:j)*64 + c]
__global__ void attn_scatter(const float* __restrict__ s, const float* __restrict__ z,
                             const int* __restrict__ seg, const int* __restrict__ gat,
                             const float* __restrict__ h, float* __restrict__ out,
                             int ldo, int ooff, int n) {
    long long t = (long long)blockIdx.x * blockDim.x + threadIdx.x;
    int j = (int)(t >> 6), c = (int)(t & 63);
    if (j >= n) return;
    int d = seg[j];
    int g = gat ? gat[j] : j;
    float a = s[j] / (z[d] + 1e-16f);
    atomicAdd(&out[(size_t)d * ldo + ooff + c], a * h[(size_t)g * 64 + c]);
}

// edge-layer node embedding: 2-endpoint softmax, written directly (no atomics)
__global__ void edge_endpoint(const float* __restrict__ q, const float* __restrict__ k,
                              const int* __restrict__ src, const int* __restrict__ dst,
                              const float* __restrict__ hn, float* __restrict__ e1, int E) {
    long long t = (long long)blockIdx.x * blockDim.x + threadIdx.x;
    int e = (int)(t >> 6), c = (int)(t & 63);
    if (e >= E) return;
    int si = src[e], di = dst[e];
    float qq = q[e];
    float ss = lrelu(qq + k[si]);
    float sd = lrelu(qq + k[di]);
    float m  = fmaxf(ss, sd);
    float es = __expf(ss - m), ed = __expf(sd - m);
    float iz = 1.f / (es + ed);
    e1[(size_t)e * 128 + c] = (es * iz) * hn[(size_t)si * 64 + c]
                            + (ed * iz) * hn[(size_t)di * 64 + c];
}

__global__ void elu_inplace(float* __restrict__ x, long long n) {
    long long i = (long long)blockIdx.x * blockDim.x + threadIdx.x;
    if (i >= n) return;
    float v = x[i];
    x[i] = v > 0.f ? v : __expf(v) - 1.f;
}

__global__ void pool_sum(const float* __restrict__ x2, const int* __restrict__ batch,
                         float* __restrict__ gsum, int N) {
    long long t = (long long)blockIdx.x * blockDim.x + threadIdx.x;
    int n = (int)(t >> 7), c = (int)(t & 127);
    if (n >= N) return;
    atomicAdd(&gsum[(size_t)batch[n] * 128 + c], x2[(size_t)n * 128 + c]);
}

__global__ void pool_cnt(const int* __restrict__ batch, float* __restrict__ cnt, int N) {
    int n = blockIdx.x * blockDim.x + threadIdx.x;
    if (n < N) atomicAdd(&cnt[batch[n]], 1.f);
}

__global__ void readout(const float* __restrict__ gsum, const float* __restrict__ cnt,
                        const float* __restrict__ roW, const float* __restrict__ rob,
                        float* __restrict__ out) {
    int t = blockIdx.x * blockDim.x + threadIdx.x;
    if (t >= BATCH * NCLS) return;
    int b = t / NCLS, cls = t % NCLS;
    float inv = 1.f / fmaxf(cnt[b], 1.f);
    float acc = rob[cls];
    for (int c = 0; c < 128; ++c)
        acc += gsum[(size_t)b * 128 + c] * inv * roW[(size_t)c * NCLS + cls];
    out[(size_t)b * NCLS + cls] = acc;
}

// ---------------------------------------------------------------------------
// Host launcher
// ---------------------------------------------------------------------------
static inline unsigned cdiv(long long a, long long b) { return (unsigned)((a + b - 1) / b); }

extern "C" void kernel_launch(void* const* d_in, const int* in_sizes, int n_in,
                              void* d_out, int out_size, void* d_ws, size_t ws_size,
                              hipStream_t stream) {
    const float* x    = (const float*)d_in[0];   // N x 128
    const float* ea   = (const float*)d_in[1];   // E x 64
    const int*   eidx = (const int*)d_in[2];     // 2 x E
    const int*   bat  = (const int*)d_in[3];     // N
    const int*   e2e  = (const int*)d_in[4];     // 2 x 3E
    const float* n1_Wn = (const float*)d_in[5];
    const float* n1_We = (const float*)d_in[6];
    const float* n1_ann = (const float*)d_in[7];
    const float* n1_ane = (const float*)d_in[8];
    const float* e1_Wn = (const float*)d_in[9];
    const float* e1_We = (const float*)d_in[10];
    const float* e1_aen = (const float*)d_in[11];
    const float* e1_aee = (const float*)d_in[12];
    const float* n2_Wn = (const float*)d_in[13];
    const float* n2_We = (const float*)d_in[14];
    const float* n2_ann = (const float*)d_in[15];
    const float* n2_ane = (const float*)d_in[16];
    const float* roW = (const float*)d_in[17];
    const float* rob = (const float*)d_in[18];
    float* out = (float*)d_out;

    const int* src  = eidx;
    const int* dst  = eidx + NE;
    const int* es_i = e2e;
    const int* ed_i = e2e + NE3;

    // --- bump-allocate workspace ---
    char* ws = (char*)d_ws;
    size_t off = 0;
    auto alloc = [&](size_t bytes) { void* p = ws + off; off = (off + bytes + 255) & ~(size_t)255; return p; };
    float*    hn   = (float*)alloc((size_t)NN  * 64  * 4);
    float*    he   = (float*)alloc((size_t)NE  * 64  * 4);
    float*    x1   = (float*)alloc((size_t)NN  * 128 * 4);
    float*    e1   = (float*)alloc((size_t)NE  * 128 * 4);
    float*    x2   = (float*)alloc((size_t)NN  * 128 * 4);
    float*    qb   = (float*)alloc((size_t)NE  * 4);
    float*    kb   = (float*)alloc((size_t)NE  * 4);
    float*    sb   = (float*)alloc((size_t)NE3 * 4);
    unsigned* menc = (unsigned*)alloc((size_t)NE * 4);
    float*    zb   = (float*)alloc((size_t)NE * 4);
    float*    gsum = (float*)alloc((size_t)BATCH * 128 * 4);
    float*    cnt  = (float*)alloc((size_t)BATCH * 4);

    const dim3 B32(32), B256(256);

    // ======================= node layer 1 =======================
    gemm_mx64_bf16<<<NN / 16, B32, 0, stream>>>(x,  n1_Wn, hn, 128, 64, 0);
    gemm_mx64_bf16<<<NE / 16, B32, 0, stream>>>(ea, n1_We, he, 64,  64, 0);
    hipMemsetAsync(x1, 0, (size_t)NN * 128 * 4, stream);
    // attention over neighbor nodes (per dst)
    dot64<<<cdiv(NN, 256), B256, 0, stream>>>(hn, n1_ann,      qb, NN);
    dot64<<<cdiv(NN, 256), B256, 0, stream>>>(hn, n1_ann + 64, kb, NN);
    fill_u32<<<cdiv(NN, 256), B256, 0, stream>>>(menc, ENC_NEGINF, NN);
    hipMemsetAsync(zb, 0, (size_t)NN * 4, stream);
    score_max<<<cdiv(NE, 256), B256, 0, stream>>>(qb, kb, dst, src, sb, menc, NE);
    exp_sum<<<cdiv(NE, 256), B256, 0, stream>>>(sb, menc, dst, zb, NE);
    attn_scatter<<<cdiv((long long)NE * 64, 256), B256, 0, stream>>>(sb, zb, dst, src, hn, x1, 128, 0, NE);
    // attention over incident edges (per dst)
    dot64<<<cdiv(NN, 256), B256, 0, stream>>>(hn, n1_ane,      qb, NN);
    dot64<<<cdiv(NE, 256), B256, 0, stream>>>(he, n1_ane + 64, kb, NE);
    fill_u32<<<cdiv(NN, 256), B256, 0, stream>>>(menc, ENC_NEGINF, NN);
    hipMemsetAsync(zb, 0, (size_t)NN * 4, stream);
    score_max<<<cdiv(NE, 256), B256, 0, stream>>>(qb, kb, dst, nullptr, sb, menc, NE);
    exp_sum<<<cdiv(NE, 256), B256, 0, stream>>>(sb, menc, dst, zb, NE);
    attn_scatter<<<cdiv((long long)NE * 64, 256), B256, 0, stream>>>(sb, zb, dst, nullptr, he, x1, 128, 64, NE);
    elu_inplace<<<cdiv((long long)NN * 128, 256), B256, 0, stream>>>(x1, (long long)NN * 128);

    // ======================= edge layer =======================
    gemm_mx64_bf16<<<NN / 16, B32, 0, stream>>>(x1, e1_Wn, hn, 128, 64, 0);
    gemm_mx64_bf16<<<NE / 16, B32, 0, stream>>>(ea, e1_We, he, 64,  64, 0);
    hipMemsetAsync(e1, 0, (size_t)NE * 128 * 4, stream);
    // endpoint (2-way) softmax embedding -> e1[:, :64]
    dot64<<<cdiv(NE, 256), B256, 0, stream>>>(he, e1_aen,      qb, NE);
    dot64<<<cdiv(NN, 256), B256, 0, stream>>>(hn, e1_aen + 64, kb, NN);
    edge_endpoint<<<cdiv((long long)NE * 64, 256), B256, 0, stream>>>(qb, kb, src, dst, hn, e1, NE);
    // adjacent-edge attention via e2e -> e1[:, 64:]
    dot64<<<cdiv(NE, 256), B256, 0, stream>>>(he, e1_aee,      qb, NE);
    dot64<<<cdiv(NE, 256), B256, 0, stream>>>(he, e1_aee + 64, kb, NE);
    fill_u32<<<cdiv(NE, 256), B256, 0, stream>>>(menc, ENC_NEGINF, NE);
    hipMemsetAsync(zb, 0, (size_t)NE * 4, stream);
    score_max<<<cdiv(NE3, 256), B256, 0, stream>>>(qb, kb, ed_i, es_i, sb, menc, NE3);
    exp_sum<<<cdiv(NE3, 256), B256, 0, stream>>>(sb, menc, ed_i, zb, NE3);
    attn_scatter<<<cdiv((long long)NE3 * 64, 256), B256, 0, stream>>>(sb, zb, ed_i, es_i, he, e1, 128, 64, NE3);
    elu_inplace<<<cdiv((long long)NE * 128, 256), B256, 0, stream>>>(e1, (long long)NE * 128);

    // ======================= node layer 2 =======================
    gemm_mx64_bf16<<<NN / 16, B32, 0, stream>>>(x1, n2_Wn, hn, 128, 64, 0);
    gemm_mx64_bf16<<<NE / 16, B32, 0, stream>>>(e1, n2_We, he, 128, 64, 0);
    hipMemsetAsync(x2, 0, (size_t)NN * 128 * 4, stream);
    dot64<<<cdiv(NN, 256), B256, 0, stream>>>(hn, n2_ann,      qb, NN);
    dot64<<<cdiv(NN, 256), B256, 0, stream>>>(hn, n2_ann + 64, kb, NN);
    fill_u32<<<cdiv(NN, 256), B256, 0, stream>>>(menc, ENC_NEGINF, NN);
    hipMemsetAsync(zb, 0, (size_t)NN * 4, stream);
    score_max<<<cdiv(NE, 256), B256, 0, stream>>>(qb, kb, dst, src, sb, menc, NE);
    exp_sum<<<cdiv(NE, 256), B256, 0, stream>>>(sb, menc, dst, zb, NE);
    attn_scatter<<<cdiv((long long)NE * 64, 256), B256, 0, stream>>>(sb, zb, dst, src, hn, x2, 128, 0, NE);
    dot64<<<cdiv(NN, 256), B256, 0, stream>>>(hn, n2_ane,      qb, NN);
    dot64<<<cdiv(NE, 256), B256, 0, stream>>>(he, n2_ane + 64, kb, NE);
    fill_u32<<<cdiv(NN, 256), B256, 0, stream>>>(menc, ENC_NEGINF, NN);
    hipMemsetAsync(zb, 0, (size_t)NN * 4, stream);
    score_max<<<cdiv(NE, 256), B256, 0, stream>>>(qb, kb, dst, nullptr, sb, menc, NE);
    exp_sum<<<cdiv(NE, 256), B256, 0, stream>>>(sb, menc, dst, zb, NE);
    attn_scatter<<<cdiv((long long)NE * 64, 256), B256, 0, stream>>>(sb, zb, dst, nullptr, he, x2, 128, 64, NE);
    elu_inplace<<<cdiv((long long)NN * 128, 256), B256, 0, stream>>>(x2, (long long)NN * 128);

    // ======================= pool + readout =======================
    hipMemsetAsync(gsum, 0, (size_t)BATCH * 128 * 4, stream);
    hipMemsetAsync(cnt,  0, (size_t)BATCH * 4, stream);
    pool_sum<<<cdiv((long long)NN * 128, 256), B256, 0, stream>>>(x2, bat, gsum, NN);
    pool_cnt<<<cdiv(NN, 256), B256, 0, stream>>>(bat, cnt, NN);
    readout<<<cdiv(BATCH * NCLS, 256), B256, 0, stream>>>(gsum, cnt, roW, rob, out);
}
